// BiRNN_CRF_27169963114547
// MI455X (gfx1250) — compile-verified, matrix operand order checked by reading, not run on hardware
//
#include <hip/hip_runtime.h>
#include <hip/hip_bf16.h>

// ---------------------------------------------------------------------------
// BiLSTM(2 layers, bidirectional) + Linear + CRF NLL for gfx1250 (CDNA5).
// B=64 T=256 E=768 H=384 4H=1536 2H=768 K=9, wave32, WMMA bf16 16x16x32.
// Recurrent kernel stages W_hh (once) and h_state (per step) into LDS via
// CDNA5 async-to-LDS instructions; WMMA fragments then come from ds loads.
// ---------------------------------------------------------------------------

#define BB   64
#define TT   256
#define EE   768
#define HH   384
#define H4   1536
#define H2   768
#define KK   9
#define MROWS (BB * TT)          // 16384

// LDS layout for lstm_rec (dynamic shared memory, byte offsets)
#define ROWP      392            // padded row stride in ushorts (784 B, 4 mod 64 banks)
#define SMW_BYTES (64 * ROWP * 2)        // 50176: W_hh slice [64 rows][384(+pad)]
#define SMH_BYTES (64 * ROWP * 2)        // 50176: h_state   [64 rows][384(+pad)]
#define SMG_BYTES (64 * 65 * 4)          // 16640: fp32 gate buffer [64][65]
#define SM_TOTAL  (SMW_BYTES + SMH_BYTES + SMG_BYTES)   // 116992 B

typedef __attribute__((ext_vector_type(16))) __bf16   v16bf;
typedef __attribute__((ext_vector_type(8)))  float    v8f;
typedef __attribute__((ext_vector_type(8)))  unsigned v8u;

__device__ __forceinline__ unsigned short f2bf(float f) {
    unsigned u = __float_as_uint(f);
    unsigned r = u + 0x7FFFu + ((u >> 16) & 1u);   // round-to-nearest-even
    return (unsigned short)(r >> 16);
}
__device__ __forceinline__ float bf2f(unsigned short u) {
    return __uint_as_float(((unsigned)u) << 16);
}
__device__ __forceinline__ float sigf(float x) {
    return 1.0f / (1.0f + __expf(-x));
}
// K-pair base for 16-bit A/B WMMA VGPR j (CDNA5 ISA 7.12.2 layout)
__device__ __forceinline__ int kbase(int j, int khalf) {
    return ((j < 4) ? (2 * j) : (16 + 2 * (j - 4))) + khalf * 8;
}

// --- CDNA5 async copy global->LDS, 16 bytes per lane (ASYNCcnt tracked) ----
__device__ __forceinline__ void async_ld16(void* lds, const void* gmem) {
    asm volatile("global_load_async_to_lds_b128 %0, %1, off"
                 :: "v"((unsigned)(size_t)lds), "v"(gmem)
                 : "memory");
}
__device__ __forceinline__ void wait_async0() {
    asm volatile("s_wait_asynccnt 0x0" ::: "memory");
}

// Build a 16x32 bf16 WMMA fragment from an LDS row (two ds_load_b128)
__device__ __forceinline__ v16bf lds_frag(const unsigned short* row, int kk, int khalf) {
    const uint4 lo = *(const uint4*)(row + kk + khalf * 8);
    const uint4 hi = *(const uint4*)(row + kk + 16 + khalf * 8);
    v8u u;
    u[0] = lo.x; u[1] = lo.y; u[2] = lo.z; u[3] = lo.w;
    u[4] = hi.x; u[5] = hi.y; u[6] = hi.z; u[7] = hi.w;
    return __builtin_bit_cast(v16bf, u);
}

// ---------------------------------------------------------------------------
// init: zero the grid-barrier counters
// ---------------------------------------------------------------------------
__global__ void init_kernel(unsigned* bars) {
    if (threadIdx.x < 8) bars[threadIdx.x] = 0u;
}

// ---------------------------------------------------------------------------
// fp32 -> bf16 convert (grid-stride)
// ---------------------------------------------------------------------------
__global__ void cvt_kernel(const float* __restrict__ src,
                           unsigned short* __restrict__ dst, int n) {
    for (int i = blockIdx.x * blockDim.x + threadIdx.x; i < n;
         i += gridDim.x * blockDim.x)
        dst[i] = f2bf(src[i]);
}

// ---------------------------------------------------------------------------
// C[M,N] = A[M,Kd] @ W[N,Kd]^T + bias[N]   (A,W bf16 row-major, C fp32)
// One wave computes a 16x64 strip (4 N-tiles), K-loop in steps of 32.
// grid: (M/16)*(N/64)/8 blocks of 256 threads (8 waves).
// ---------------------------------------------------------------------------
__global__ void wmma_gemm_bias(const unsigned short* __restrict__ A,
                               const unsigned short* __restrict__ W,
                               const float* __restrict__ bias,
                               float* __restrict__ C,
                               int M, int N, int Kd) {
    const int lane  = threadIdx.x & 31;
    const int l15   = lane & 15;
    const int khalf = lane >> 4;
    const int wgid  = blockIdx.x * 8 + (threadIdx.x >> 5);
    const int strips = N >> 6;                 // N/64
    const int rt = wgid / strips;              // row tile
    const int cs = wgid % strips;              // 64-wide col strip
    if (rt * 16 >= M) return;

    const int mrow = rt * 16 + l15;
    v8f acc0 = {}, acc1 = {}, acc2 = {}, acc3 = {};

    for (int kk = 0; kk < Kd; kk += 32) {
        v8u au;
        const unsigned short* arow = A + (size_t)mrow * Kd + kk;
#pragma unroll
        for (int j = 0; j < 8; ++j)
            au[j] = *(const unsigned*)(arow + kbase(j, khalf));
        v16bf a = __builtin_bit_cast(v16bf, au);

#pragma unroll
        for (int s = 0; s < 4; ++s) {
            const int ncol = cs * 64 + s * 16 + l15;
            const unsigned short* wrow = W + (size_t)ncol * Kd + kk;
            v8u bu;
#pragma unroll
            for (int j = 0; j < 8; ++j)
                bu[j] = *(const unsigned*)(wrow + kbase(j, khalf));
            v16bf b = __builtin_bit_cast(v16bf, bu);
            if (s == 0) acc0 = __builtin_amdgcn_wmma_f32_16x16x32_bf16(false, a, false, b, (short)0, acc0, false, false);
            if (s == 1) acc1 = __builtin_amdgcn_wmma_f32_16x16x32_bf16(false, a, false, b, (short)0, acc1, false, false);
            if (s == 2) acc2 = __builtin_amdgcn_wmma_f32_16x16x32_bf16(false, a, false, b, (short)0, acc2, false, false);
            if (s == 3) acc3 = __builtin_amdgcn_wmma_f32_16x16x32_bf16(false, a, false, b, (short)0, acc3, false, false);
        }
    }

#pragma unroll
    for (int s = 0; s < 4; ++s) {
        const int ncol = cs * 64 + s * 16 + l15;
        const float bv = bias[ncol];
        v8f acc = (s == 0) ? acc0 : (s == 1) ? acc1 : (s == 2) ? acc2 : acc3;
#pragma unroll
        for (int v = 0; v < 8; ++v) {
            const int m = rt * 16 + v + 8 * khalf;   // C/D layout: M = v + 8*(lane>=16)
            C[(size_t)m * N + ncol] = acc[v] + bv;
        }
    }
}

// ---------------------------------------------------------------------------
// Persistent bidirectional LSTM recurrence for one layer.
// grid = 48 blocks: [0,24) forward dir, [24,48) backward dir.
// Block owns hidden slice j0 = 16*(blockIdx%24).
//   prologue: async-stage W_hh slice [64 x 384] bf16 -> LDS (reused 256 steps)
//   per step: async-stage h_state [64 x 384] bf16 -> LDS,
//             16 WMMA tiles (4 gates x 4 batch tiles) over 8 waves with
//             A/B fragments from ds_load_b128, gate math via LDS fp32 buffer,
//             c(f32)/h(bf16) state update, h into [B,T,2H] concat output,
//             grid barrier (monotonic atomic counter + s_sleep spin).
// ---------------------------------------------------------------------------
__global__ void lstm_rec_kernel(const float* __restrict__ xg,           // [2][B,T,4H]
                                const unsigned short* __restrict__ whh, // [2][4H,H]
                                unsigned short* __restrict__ h_state,   // [2][B,H] bf16
                                float* __restrict__ c_state,            // [2][B,H]
                                unsigned short* __restrict__ h_out,     // [B,T,2H] bf16
                                unsigned* __restrict__ bar) {
    extern __shared__ __align__(16) char smem[];
    unsigned short* sm_w = (unsigned short*)smem;                  // [64][ROWP]
    unsigned short* sm_h = (unsigned short*)(smem + SMW_BYTES);    // [64][ROWP]
    float (*gbuf)[65]    = (float(*)[65])(smem + SMW_BYTES + SMH_BYTES);

    const int dir = blockIdx.x / 24;           // 0 = fwd, 1 = bwd
    const int j0  = (blockIdx.x % 24) * 16;    // hidden-unit slice

    const float*          xg_d = xg + (size_t)dir * MROWS * H4;
    const unsigned short* w    = whh + (size_t)dir * H4 * HH;
    unsigned short*       hs   = h_state + (size_t)dir * BB * HH;
    float*                cs   = c_state + (size_t)dir * BB * HH;
    const int col_off = dir * HH;

    const int lane  = threadIdx.x & 31;
    const int l15   = lane & 15;
    const int khalf = lane >> 4;
    const int wid   = threadIdx.x >> 5;

    // ---- stage W_hh slice into LDS once (rows: gate*16 + r <-> w row gate*H+j0+r)
    // 64 rows x 48 16-byte transfers per row = 3072 transfers, 12 per thread
    for (int i = threadIdx.x; i < 64 * 48; i += 256) {
        const int row = i / 48, off = i % 48;
        const int gate = row >> 4, r = row & 15;
        const char* src = (const char*)(w + (size_t)(gate * HH + j0 + r) * HH) + off * 16;
        async_ld16((char*)(sm_w + row * ROWP) + off * 16, src);
    }
    wait_async0();
    __syncthreads();

    for (int t = 0; t < TT; ++t) {
        const int teff = dir ? (TT - 1 - t) : t;

        // ---- stage h_state (previous step, written by all blocks) into LDS
        if (t > 0) {
            for (int i = threadIdx.x; i < 64 * 48; i += 256) {
                const int row = i / 48, off = i % 48;
                async_ld16((char*)(sm_h + row * ROWP) + off * 16,
                           (const char*)hs + (size_t)i * 16);
            }
            wait_async0();
        }
        __syncthreads();

#pragma unroll
        for (int half = 0; half < 2; ++half) {
            const int tile = wid + half * 8;   // 0..15
            const int gate = tile >> 2;        // 0..3 (i,f,g,o)
            const int rt   = tile & 3;         // batch tile 0..3
            const int ncol = gate * HH + j0 + l15;   // column in 4H

            v8f acc = {};
            if (t > 0) {
                const unsigned short* arow = sm_h + (rt * 16 + l15) * ROWP;
                const unsigned short* wrow = sm_w + (gate * 16 + l15) * ROWP;
                for (int kk = 0; kk < HH; kk += 32) {
                    v16bf a = lds_frag(arow, kk, khalf);
                    v16bf b = lds_frag(wrow, kk, khalf);
                    acc = __builtin_amdgcn_wmma_f32_16x16x32_bf16(
                        false, a, false, b, (short)0, acc, false, false);
                }
            }
#pragma unroll
            for (int v = 0; v < 8; ++v) {
                const int b = rt * 16 + v + 8 * khalf;
                gbuf[b][gate * 16 + l15] =
                    acc[v] + xg_d[((size_t)b * TT + teff) * H4 + ncol];
            }
        }
        __syncthreads();

        // gate nonlinearity + state update: 64 batches x 16 units
        for (int e = threadIdx.x; e < 64 * 16; e += 256) {
            const int b  = e >> 4;
            const int jl = e & 15;
            const float gi = gbuf[b][jl];
            const float gf = gbuf[b][16 + jl];
            const float gg = gbuf[b][32 + jl];
            const float go = gbuf[b][48 + jl];
            const int hidx = b * HH + j0 + jl;
            const float cold = (t == 0) ? 0.0f : cs[hidx];
            const float cnew = sigf(gf) * cold + sigf(gi) * tanhf(gg);
            const float h    = sigf(go) * tanhf(cnew);
            cs[hidx] = cnew;
            const unsigned short hb = f2bf(h);
            hs[hidx] = hb;
            h_out[((size_t)b * TT + teff) * H2 + col_off + j0 + jl] = hb;
        }
        __threadfence();
        __syncthreads();

        // grid-wide barrier between timesteps (monotonic counter)
        if (threadIdx.x == 0) {
            const unsigned target = (unsigned)(t + 1) * 48u;
            atomicAdd(bar, 1u);
            while (__hip_atomic_load(bar, __ATOMIC_ACQUIRE,
                                     __HIP_MEMORY_SCOPE_AGENT) < target)
                __builtin_amdgcn_s_sleep(2);
        }
        __syncthreads();
    }
}

// ---------------------------------------------------------------------------
// emissions[B*T, 9] = h1(bf16)[B*T, 768] @ w_out[9,768]^T + b_out
// ---------------------------------------------------------------------------
__global__ void emissions_kernel(const unsigned short* __restrict__ h1,
                                 const float* __restrict__ w_out,
                                 const float* __restrict__ b_out,
                                 float* __restrict__ em) {
    const int idx = blockIdx.x * blockDim.x + threadIdx.x;
    if (idx >= MROWS * KK) return;
    const int k   = idx % KK;
    const int row = idx / KK;
    const unsigned short* hrow = h1 + (size_t)row * H2;
    const float* wrow = w_out + (size_t)k * H2;
    float s = b_out[k];
    for (int j = 0; j < H2; ++j) s += bf2f(hrow[j]) * wrow[j];
    em[idx] = s;
}

// ---------------------------------------------------------------------------
// CRF NLL, full mask, reduction='sum'. One thread per batch, single block.
// ---------------------------------------------------------------------------
__global__ void crf_kernel(const float* __restrict__ em,   // [B,T,K]
                           const int* __restrict__ tags,   // [B,T]
                           const float* __restrict__ st,
                           const float* __restrict__ et,
                           const float* __restrict__ tr,   // [K,K]
                           float* __restrict__ out) {
    __shared__ float partial[64];
    const int b = threadIdx.x;
    if (b < BB) {
        const float* e  = em + (size_t)b * TT * KK;
        const int*   tg = tags + (size_t)b * TT;
        float alpha[KK], na[KK];
        int prev = tg[0];
        float num = st[prev] + e[prev];
        for (int k = 0; k < KK; ++k) alpha[k] = st[k] + e[k];
        for (int t = 1; t < TT; ++t) {
            const int cur = tg[t];
            num += e[t * KK + cur] + tr[prev * KK + cur];
            prev = cur;
            for (int k = 0; k < KK; ++k) {
                float m = -3.4e38f;
                for (int j = 0; j < KK; ++j)
                    m = fmaxf(m, alpha[j] + tr[j * KK + k]);
                float s = 0.0f;
                for (int j = 0; j < KK; ++j)
                    s += __expf(alpha[j] + tr[j * KK + k] - m);
                na[k] = m + __logf(s) + e[t * KK + k];
            }
            for (int k = 0; k < KK; ++k) alpha[k] = na[k];
        }
        num += et[prev];
        float m = -3.4e38f;
        for (int k = 0; k < KK; ++k) m = fmaxf(m, alpha[k] + et[k]);
        float s = 0.0f;
        for (int k = 0; k < KK; ++k) s += __expf(alpha[k] + et[k] - m);
        const float den = m + __logf(s);
        partial[b] = den - num;                 // -(num - den)
    }
    __syncthreads();
    if (threadIdx.x == 0) {
        float tot = 0.0f;
        for (int i = 0; i < BB; ++i) tot += partial[i];
        out[0] = tot;
    }
}

// ---------------------------------------------------------------------------
// host-side orchestration
// ---------------------------------------------------------------------------
extern "C" void kernel_launch(void* const* d_in, const int* in_sizes, int n_in,
                              void* d_out, int out_size, void* d_ws, size_t ws_size,
                              hipStream_t stream) {
    (void)in_sizes; (void)n_in; (void)out_size; (void)ws_size;

    const float* emb   = (const float*)d_in[0];
    const int*   tags  = (const int*)d_in[1];
    // d_in[2] attention mask: full mask in the reference -> unused
    const float* wih_f32[4] = { (const float*)d_in[3],  (const float*)d_in[6],
                                (const float*)d_in[9],  (const float*)d_in[12] };
    const float* whh_f32[4] = { (const float*)d_in[4],  (const float*)d_in[7],
                                (const float*)d_in[10], (const float*)d_in[13] };
    const float* bias[4]    = { (const float*)d_in[5],  (const float*)d_in[8],
                                (const float*)d_in[11], (const float*)d_in[14] };
    const float* w_out = (const float*)d_in[15];
    const float* b_out = (const float*)d_in[16];
    const float* st    = (const float*)d_in[17];
    const float* et    = (const float*)d_in[18];
    const float* tr    = (const float*)d_in[19];

    char* ws = (char*)d_ws;
    size_t off = 0;
    auto alloc = [&](size_t bytes) { size_t o = off; off = (off + bytes + 255) & ~(size_t)255; return o; };

    // buf0: x bf16, later reused as h0 concat bf16 (both B*T*768 elems)
    unsigned short* buf0 = (unsigned short*)(ws + alloc((size_t)MROWS * EE * 2));
    unsigned short* buf1 = (unsigned short*)(ws + alloc((size_t)MROWS * H2 * 2)); // h1 concat
    unsigned short* wih  = (unsigned short*)(ws + alloc((size_t)4 * H4 * EE * 2));
    unsigned short* whh  = (unsigned short*)(ws + alloc((size_t)4 * H4 * HH * 2));
    float*          xg   = (float*)(ws + alloc((size_t)2 * MROWS * H4 * 4));
    unsigned short* hst  = (unsigned short*)(ws + alloc((size_t)2 * BB * HH * 2));
    float*          cst  = (float*)(ws + alloc((size_t)2 * BB * HH * 4));
    float*          em   = (float*)(ws + alloc((size_t)MROWS * KK * 4));
    unsigned*       bars = (unsigned*)(ws + alloc(256));

    init_kernel<<<1, 64, 0, stream>>>(bars);

    // bf16 conversions
    cvt_kernel<<<4096, 256, 0, stream>>>(emb, buf0, MROWS * EE);
    for (int d = 0; d < 4; ++d) {
        cvt_kernel<<<1024, 256, 0, stream>>>(wih_f32[d], wih + (size_t)d * H4 * EE, H4 * EE);
        cvt_kernel<<<512,  256, 0, stream>>>(whh_f32[d], whh + (size_t)d * H4 * HH, H4 * HH);
    }

    const int gemm_blocks = (MROWS / 16) * (H4 / 64) / 8;   // 3072

    // ---- layer 0: input projections + recurrence ----
    wmma_gemm_bias<<<gemm_blocks, 256, 0, stream>>>(
        buf0, wih + 0 * (size_t)H4 * EE, bias[0], xg, MROWS, H4, EE);
    wmma_gemm_bias<<<gemm_blocks, 256, 0, stream>>>(
        buf0, wih + 1 * (size_t)H4 * EE, bias[1], xg + (size_t)MROWS * H4, MROWS, H4, EE);
    lstm_rec_kernel<<<48, 256, SM_TOTAL, stream>>>(
        xg, whh + 0 * (size_t)H4 * HH, hst, cst, buf0 /* h0 overwrites x */, bars + 0);

    // ---- layer 1: input projections + recurrence ----
    wmma_gemm_bias<<<gemm_blocks, 256, 0, stream>>>(
        buf0, wih + 2 * (size_t)H4 * EE, bias[2], xg, MROWS, H4, H2);
    wmma_gemm_bias<<<gemm_blocks, 256, 0, stream>>>(
        buf0, wih + 3 * (size_t)H4 * EE, bias[3], xg + (size_t)MROWS * H4, MROWS, H4, H2);
    lstm_rec_kernel<<<48, 256, SM_TOTAL, stream>>>(
        xg, whh + 2 * (size_t)H4 * HH, hst, cst, buf1, bars + 1);

    // ---- emissions + CRF ----
    emissions_kernel<<<(MROWS * KK + 255) / 256, 256, 0, stream>>>(buf1, w_out, b_out, em);
    crf_kernel<<<1, 64, 0, stream>>>(em, tags, st, et, tr, (float*)d_out);
}
